// Discriminator_2138893713484
// MI455X (gfx1250) — compile-verified, matrix operand order checked by reading, not run on hardware
//
#include <hip/hip_runtime.h>
#include <hip/hip_bf16.h>

// Problem constants (match reference)
#define B_  32
#define S_  64
#define T_  64
#define H_  256
#define H3_ 768
#define V_  32000
#define SOS 1

typedef __attribute__((ext_vector_type(16))) _Float16 v16h;
typedef __attribute__((ext_vector_type(8)))  _Float16 v8h;
typedef __attribute__((ext_vector_type(8)))  float    v8f;

// ---------------------------------------------------------------------------
// WMMA helpers (CDNA5 16x16x32 f16 -> f32)
// A fragment (16x32, row-major source, leading dim ld):
//   element i of lane l holds (m,k) = (row0 + l%16, k0 + (i/8)*16 + (l/16)*8 + i%8)
// B fragment for D = A x B with B = W^T (W stored row-major [N][K]) uses the
// identical mapping applied to W (i.e. A-layout of B^T).
// ---------------------------------------------------------------------------
__device__ __forceinline__ v16h load_frag(const _Float16* p, int ld, int row0, int k0) {
  const int lane = threadIdx.x & 31;
  const _Float16* base = p + (row0 + (lane & 15)) * ld + k0 + ((lane >> 4) << 3);
  v8h lo = *(const v8h*)(base);        // k = k0 + (l/16)*8 + 0..7
  v8h hi = *(const v8h*)(base + 16);   // k = k0 + 16 + (l/16)*8 + 0..7
  v16h r;
#pragma unroll
  for (int i = 0; i < 8; ++i) { r[i] = lo[i]; r[i + 8] = hi[i]; }
  return r;
}

__device__ __forceinline__ v8f wmma16(v16h a, v16h b, v8f c) {
  // (neg_a, A, neg_b, B, c_mod, C, reuse_a, reuse_b)
  return __builtin_amdgcn_wmma_f32_16x16x32_f16(false, a, false, b, (short)0, c,
                                                false, false);
}

__device__ __forceinline__ float sigmoidf_(float x) {
  return 1.0f / (1.0f + __expf(-x));
}

// ---------------------------------------------------------------------------
// Prep kernels
// ---------------------------------------------------------------------------
__global__ void cvt_f16_kernel(_Float16* dst, const float* src, int n) {
  for (int i = blockIdx.x * blockDim.x + threadIdx.x; i < n;
       i += gridDim.x * blockDim.x)
    dst[i] = (_Float16)src[i];
}

// Gather encoder inputs for ALL steps: X[(s*B+b)][k] = enc_emb[context[b][s]][k]
__global__ void gather_enc_x_kernel(_Float16* xh, const int* ctx_tok,
                                    const float* emb) {
  const int n = S_ * B_ * H_;
  for (int i = blockIdx.x * blockDim.x + threadIdx.x; i < n;
       i += gridDim.x * blockDim.x) {
    int row = i >> 8, k = i & (H_ - 1);
    int s = row >> 5, b = row & (B_ - 1);
    int tok = ctx_tok[b * S_ + s];
    xh[i] = (_Float16)emb[tok * H_ + k];
  }
}

// Step-invariant decoder combiner bias: comb_sos[j] = b_comb[j] + emb_sos . Wc1[j]
__global__ void comb_sos_kernel(float* comb_sos, const float* dec_emb,
                                const float* Wc, const float* b_comb) {
  int j = threadIdx.x;  // 256 threads
  float acc = b_comb[j];
  for (int k = 0; k < H_; ++k)
    acc += dec_emb[SOS * H_ + k] * Wc[j * (2 * H_) + k];
  comb_sos[j] = acc;
}

// ---------------------------------------------------------------------------
// Encoder input-gate GEMM (fully parallel, hoisted out of the recurrence):
// Gi[2048][768] = X[2048][256] x enc_Wih^T   (bias added later, elementwise)
// One wave per 16x16 output tile; 128 x 48 tiles; 8 waves/block -> 768 blocks.
// ---------------------------------------------------------------------------
__global__ void __launch_bounds__(256) enc_gi_gemm(float* Gi, const _Float16* X,
                                                   const _Float16* Wih) {
  const int wid = (blockIdx.x * blockDim.x + threadIdx.x) >> 5;
  const int mt = wid / 48, nt = wid % 48;
  v8f acc = {};
#pragma unroll
  for (int kt = 0; kt < 8; ++kt) {
    v16h a = load_frag(X, H_, mt * 16, kt * 32);
    v16h b = load_frag(Wih, H_, nt * 16, kt * 32);
    acc = wmma16(a, b, acc);
  }
  const int lane = threadIdx.x & 31;
  const int m0 = mt * 16 + ((lane >> 4) << 3);
  const int n = nt * 16 + (lane & 15);
#pragma unroll
  for (int r = 0; r < 8; ++r) Gi[(m0 + r) * H3_ + n] = acc[r];
}

// ---------------------------------------------------------------------------
// Encoder recurrence: one 1024-thread workgroup (32 wave32 = full WGP).
// Per step: gh = h x Whh^T via WMMA (96 tiles / 32 waves), then GRU update.
// h kept resident in LDS (f16 operand copy + f32 state).
// ---------------------------------------------------------------------------
__global__ void __launch_bounds__(1024) enc_seq_kernel(
    float* enc_outs, float* gh_ws, const float* Gi, const _Float16* Whh,
    const float* bih, const float* bhh) {
  __shared__ _Float16 h_h[B_ * H_];  // 16 KB : WMMA A operand
  __shared__ float h_f[B_ * H_];     // 32 KB : f32 recurrent state
  const int tid = threadIdx.x;
  const int wave = tid >> 5;
  const int lane = tid & 31;

  for (int e = tid; e < B_ * H_; e += 1024) {
    h_h[e] = (_Float16)0.0f;
    h_f[e] = 0.0f;
  }
  __syncthreads();

  for (int s = 0; s < S_; ++s) {
    // ---- gh = h @ Whh^T : M=32 (2 tiles) x N=768 (48 tiles), 3 tiles/wave
#pragma unroll
    for (int tt = 0; tt < 3; ++tt) {
      const int tile = wave * 3 + tt;
      const int mt = tile / 48, nt = tile % 48;
      v8f acc = {};
#pragma unroll
      for (int kt = 0; kt < 8; ++kt) {
        v16h a = load_frag(h_h, H_, mt * 16, kt * 32);
        v16h b = load_frag(Whh, H_, nt * 16, kt * 32);
        acc = wmma16(a, b, acc);
      }
      const int m0 = mt * 16 + ((lane >> 4) << 3);
      const int n = nt * 16 + (lane & 15);
#pragma unroll
      for (int r = 0; r < 8; ++r) gh_ws[(m0 + r) * H3_ + n] = acc[r];
    }
    __threadfence_block();
    __syncthreads();

    // ---- GRU elementwise update (8192 elems / 1024 threads)
#pragma unroll
    for (int it = 0; it < 8; ++it) {
      const int e = tid + it * 1024;
      const int b = e >> 8, j = e & (H_ - 1);
      const float* gi = Gi + (s * B_ + b) * H3_;
      const float* gh = gh_ws + b * H3_;
      float r = sigmoidf_(gi[j] + bih[j] + gh[j] + bhh[j]);
      float z = sigmoidf_(gi[H_ + j] + bih[H_ + j] + gh[H_ + j] + bhh[H_ + j]);
      float nn = tanhf(gi[2 * H_ + j] + bih[2 * H_ + j] +
                       r * (gh[2 * H_ + j] + bhh[2 * H_ + j]));
      float hn = (1.0f - z) * nn + z * h_f[e];
      enc_outs[(s * B_ + b) * H_ + j] = hn;
      h_f[e] = hn;
      h_h[e] = (_Float16)hn;
    }
    __threadfence_block();
    __syncthreads();
  }
}

// ---------------------------------------------------------------------------
// Decoder recurrence: attention + combiner GEMM + GRU GEMMs + gathered output
// projection (only W_out rows that are actually observed are touched).
// ---------------------------------------------------------------------------
__global__ void __launch_bounds__(1024) dec_seq_kernel(
    float* out_probs,                 // [B][T] (d_out + 1)
    float* h_fws, float* gg_ws,       // h f32 state; gg_ws = gi[32x768] | gh[32x768]
    const float* enc_outs, const float* comb_sos, const _Float16* Wcomb_h,
    const _Float16* Wih_h, const _Float16* Whh_h, const float* bih,
    const float* bhh, const int* input_tensor, const float* W_out,
    const float* b_out) {
  __shared__ _Float16 h_h[B_ * H_];    // 16 KB
  __shared__ _Float16 ctx_h[B_ * H_];  // 16 KB
  __shared__ _Float16 x_h[B_ * H_];    // 16 KB
  __shared__ float sc[B_][S_ + 4];     // ~8.7 KB, padded vs bank conflicts
  const int tid = threadIdx.x;
  const int wave = tid >> 5;
  const int lane = tid & 31;

  // init h = final encoder state (enc_outs step 63)
  for (int e = tid; e < B_ * H_; e += 1024) {
    float v = enc_outs[(S_ - 1) * B_ * H_ + e];
    h_h[e] = (_Float16)v;
    h_fws[e] = v;
  }
  __syncthreads();

  for (int t = 0; t < T_; ++t) {
    // ---- 1. attention scores: sc[b][s] = enc_outs[s,b,:] . h[b,:]
#pragma unroll
    for (int it = 0; it < 2; ++it) {
      const int e = tid + it * 1024;  // e = b*64 + s
      const int b = e >> 6, s = e & (S_ - 1);
      const float* eo = enc_outs + (s * B_ + b) * H_;
      const _Float16* hb = h_h + b * H_;
      float acc = 0.0f;
      for (int k = 0; k < H_; ++k) acc += eo[k] * (float)hb[k];
      sc[b][s] = acc;
    }
    __syncthreads();

    // ---- 2. softmax over s (one thread per batch row; 64-wide, cheap)
    if (tid < B_) {
      const int b = tid;
      float mx = sc[b][0];
      for (int s = 1; s < S_; ++s) mx = fmaxf(mx, sc[b][s]);
      float sum = 0.0f;
      for (int s = 0; s < S_; ++s) {
        float ev = __expf(sc[b][s] - mx);
        sc[b][s] = ev;
        sum += ev;
      }
      float inv = 1.0f / sum;
      for (int s = 0; s < S_; ++s) sc[b][s] *= inv;
    }
    __syncthreads();

    // ---- 3. ctx[b][j] = sum_s attn[b][s] * enc_outs[s][b][j]
#pragma unroll
    for (int it = 0; it < 8; ++it) {
      const int e = tid + it * 1024;
      const int b = e >> 8, j = e & (H_ - 1);
      float acc = 0.0f;
      for (int s = 0; s < S_; ++s) acc += sc[b][s] * enc_outs[(s * B_ + b) * H_ + j];
      ctx_h[e] = (_Float16)acc;
    }
    __syncthreads();

    // ---- 4. x = relu(comb_sos + ctx @ Wc2^T) : 2x16 tiles, one per wave
    {
      const int mt = wave >> 4, nt = wave & 15;
      v8f acc = {};
#pragma unroll
      for (int kt = 0; kt < 8; ++kt) {
        v16h a = load_frag(ctx_h, H_, mt * 16, kt * 32);
        v16h b = load_frag(Wcomb_h + H_, 2 * H_, nt * 16, kt * 32);  // Wc2 half
        acc = wmma16(a, b, acc);
      }
      const int m0 = mt * 16 + ((lane >> 4) << 3);
      const int n = nt * 16 + (lane & 15);
      const float cb = comb_sos[n];
#pragma unroll
      for (int r = 0; r < 8; ++r)
        x_h[(m0 + r) * H_ + n] = (_Float16)fmaxf(acc[r] + cb, 0.0f);
    }
    __syncthreads();

    // ---- 5. gi = x @ Wih^T and gh = h @ Whh^T : 192 tiles, 6 per wave
#pragma unroll
    for (int tt = 0; tt < 6; ++tt) {
      const int tile = wave * 6 + tt;
      const int half = tile / 96;  // 0: gi (A=x), 1: gh (A=h)
      const int tl = tile % 96;
      const int mt = tl / 48, nt = tl % 48;
      const _Float16* A = half ? h_h : x_h;
      const _Float16* W = half ? Whh_h : Wih_h;
      v8f acc = {};
#pragma unroll
      for (int kt = 0; kt < 8; ++kt) {
        v16h a = load_frag(A, H_, mt * 16, kt * 32);
        v16h b = load_frag(W, H_, nt * 16, kt * 32);
        acc = wmma16(a, b, acc);
      }
      float* dst = gg_ws + half * (B_ * H3_);
      const int m0 = mt * 16 + ((lane >> 4) << 3);
      const int n = nt * 16 + (lane & 15);
#pragma unroll
      for (int r = 0; r < 8; ++r) dst[(m0 + r) * H3_ + n] = acc[r];
    }
    __threadfence_block();
    __syncthreads();

    // ---- 6. GRU elementwise update
#pragma unroll
    for (int it = 0; it < 8; ++it) {
      const int e = tid + it * 1024;
      const int b = e >> 8, j = e & (H_ - 1);
      const float* gi = gg_ws + b * H3_;
      const float* gh = gg_ws + B_ * H3_ + b * H3_;
      float r = sigmoidf_(gi[j] + bih[j] + gh[j] + bhh[j]);
      float z = sigmoidf_(gi[H_ + j] + bih[H_ + j] + gh[H_ + j] + bhh[H_ + j]);
      float nn = tanhf(gi[2 * H_ + j] + bih[2 * H_ + j] +
                       r * (gh[2 * H_ + j] + bhh[2 * H_ + j]));
      float hn = (1.0f - z) * nn + z * h_fws[e];
      h_fws[e] = hn;
      h_h[e] = (_Float16)hn;
    }
    __threadfence_block();
    __syncthreads();

    // ---- 7. gathered output projection: p[b] = sigmoid(h . W_out[tok] + b_out[tok])
    //         (replaces the [32,256]x[256,32000] projection — only gathered
    //          rows are observable). One wave per batch row, shuffle-reduce.
    {
      const int b = wave;
      const int tok = input_tensor[b * T_ + t];
      const float* wrow = W_out + (long)tok * H_;
      float acc = 0.0f;
#pragma unroll
      for (int k = lane; k < H_; k += 32) acc += h_fws[b * H_ + k] * wrow[k];
#pragma unroll
      for (int off = 16; off > 0; off >>= 1) acc += __shfl_xor(acc, off, 32);
      if (lane == 0) out_probs[b * T_ + t] = sigmoidf_(acc + b_out[tok]);
    }
    __syncthreads();
  }
}

// ---------------------------------------------------------------------------
// BCE loss vs constant target (log clamped at -100, PyTorch BCELoss semantics)
// ---------------------------------------------------------------------------
__global__ void __launch_bounds__(1024) loss_kernel(float* d_out,
                                                    const int* true_sample) {
  __shared__ float red[1024];
  const float* p = d_out + 1;
  const float tv = (float)(*true_sample);
  const int tid = threadIdx.x;
  float acc = 0.0f;
  for (int e = tid; e < B_ * T_; e += 1024) {
    float pv = p[e];
    float lp = fmaxf(logf(pv), -100.0f);
    float l1p = fmaxf(log1pf(-pv), -100.0f);
    acc += tv * lp + (1.0f - tv) * l1p;
  }
  red[tid] = acc;
  __syncthreads();
  for (int s = 512; s > 0; s >>= 1) {
    if (tid < s) red[tid] += red[tid + s];
    __syncthreads();
  }
  if (tid == 0) d_out[0] = -red[0] / (float)(B_ * T_);
}

// ---------------------------------------------------------------------------
// Host-side launch
// ---------------------------------------------------------------------------
extern "C" void kernel_launch(void* const* d_in, const int* in_sizes, int n_in,
                              void* d_out, int out_size, void* d_ws,
                              size_t ws_size, hipStream_t stream) {
  (void)in_sizes; (void)n_in; (void)out_size; (void)ws_size;

  const int* context = (const int*)d_in[0];
  const int* input_t = (const int*)d_in[1];
  const int* true_sm = (const int*)d_in[2];
  const float* enc_emb = (const float*)d_in[3];
  const float* enc_Wih = (const float*)d_in[4];
  const float* enc_Whh = (const float*)d_in[5];
  const float* enc_bih = (const float*)d_in[6];
  const float* enc_bhh = (const float*)d_in[7];
  const float* dec_emb = (const float*)d_in[8];
  const float* W_comb = (const float*)d_in[9];
  const float* b_comb = (const float*)d_in[10];
  const float* dec_Wih = (const float*)d_in[11];
  const float* dec_Whh = (const float*)d_in[12];
  const float* dec_bih = (const float*)d_in[13];
  const float* dec_bhh = (const float*)d_in[14];
  const float* W_out = (const float*)d_in[15];
  const float* b_out = (const float*)d_in[16];
  float* out = (float*)d_out;  // [0] = loss, [1..2048] = probs [B][T]

  // Workspace carve-out (256B aligned), total ~11.2 MB
  char* w = (char*)d_ws;
  auto carve = [&](size_t bytes) {
    void* p = (void*)w;
    w += (bytes + 255) & ~(size_t)255;
    return p;
  };
  _Float16* encWih_h = (_Float16*)carve(H3_ * H_ * 2);
  _Float16* encWhh_h = (_Float16*)carve(H3_ * H_ * 2);
  _Float16* decWih_h = (_Float16*)carve(H3_ * H_ * 2);
  _Float16* decWhh_h = (_Float16*)carve(H3_ * H_ * 2);
  _Float16* Wcomb_h  = (_Float16*)carve(H_ * 2 * H_ * 2);
  _Float16* Xenc_h   = (_Float16*)carve((size_t)S_ * B_ * H_ * 2);
  float* Gi_enc   = (float*)carve((size_t)S_ * B_ * H3_ * 4);
  float* enc_outs = (float*)carve((size_t)S_ * B_ * H_ * 4);
  float* gh_ws    = (float*)carve((size_t)B_ * H3_ * 4);
  float* gg_ws    = (float*)carve((size_t)2 * B_ * H3_ * 4);
  float* comb_sos = (float*)carve(H_ * 4);
  float* h_fws    = (float*)carve((size_t)B_ * H_ * 4);

  // 1. weight conversions f32 -> f16 (parallel, off the critical path)
  cvt_f16_kernel<<<384, 256, 0, stream>>>(encWih_h, enc_Wih, H3_ * H_);
  cvt_f16_kernel<<<384, 256, 0, stream>>>(encWhh_h, enc_Whh, H3_ * H_);
  cvt_f16_kernel<<<384, 256, 0, stream>>>(decWih_h, dec_Wih, H3_ * H_);
  cvt_f16_kernel<<<384, 256, 0, stream>>>(decWhh_h, dec_Whh, H3_ * H_);
  cvt_f16_kernel<<<256, 256, 0, stream>>>(Wcomb_h, W_comb, H_ * 2 * H_);

  // 2. gather all encoder step inputs + step-invariant decoder bias
  gather_enc_x_kernel<<<512, 256, 0, stream>>>(Xenc_h, context, enc_emb);
  comb_sos_kernel<<<1, 256, 0, stream>>>(comb_sos, dec_emb, W_comb, b_comb);

  // 3. hoisted encoder input-gate GEMM: [2048,256] x [256,768] (WMMA)
  enc_gi_gemm<<<768, 256, 0, stream>>>(Gi_enc, Xenc_h, encWih_h);

  // 4. encoder recurrence (single WGP, WMMA per step)
  enc_seq_kernel<<<1, 1024, 0, stream>>>(enc_outs, gh_ws, Gi_enc, encWhh_h,
                                         enc_bih, enc_bhh);

  // 5. decoder recurrence (attention + WMMA GEMMs + gathered projection)
  dec_seq_kernel<<<1, 1024, 0, stream>>>(out + 1, h_fws, gg_ws, enc_outs,
                                         comb_sos, Wcomb_h, decWih_h, decWhh_h,
                                         dec_bih, dec_bhh, input_t, W_out,
                                         b_out);

  // 6. BCE loss reduction
  loss_kernel<<<1, 1024, 0, stream>>>(out, true_sm);
}
